// GNN_64003602645422
// MI455X (gfx1250) — matrix operand order repaired and checked by reading, not compiled, not run on hardware
//
#include <hip/hip_runtime.h>
#include <math.h>

typedef __attribute__((ext_vector_type(16))) _Float16 v16h;
typedef __attribute__((ext_vector_type(8)))  _Float16 v8h;
typedef __attribute__((ext_vector_type(8)))  float    v8f;

#define FT 128
#define NEG_SLOPE 0.2f

// ---------------------------------------------------------------------------
// utility: fill buffer with a constant
// ---------------------------------------------------------------------------
__global__ void fill_f32(float* __restrict__ p, float v, int n) {
    int i = blockIdx.x * blockDim.x + threadIdx.x;
    if (i < n) p[i] = v;
}

// convert f32 -> f16 (element-wise)
__global__ void cvt_f32_f16(const float* __restrict__ s, _Float16* __restrict__ d, int n) {
    int i = blockIdx.x * blockDim.x + threadIdx.x;
    if (i < n) d[i] = (_Float16)s[i];
}

// convert + transpose 128x128 weight: Wt[n*128+k] = (f16)W[k*128+n]  (col-major f16)
__global__ void cvt_transpose_w(const float* __restrict__ W, _Float16* __restrict__ Wt) {
    int idx = blockIdx.x * blockDim.x + threadIdx.x;
    if (idx >= FT * FT) return;
    int n = idx >> 7, k = idx & 127;
    Wt[n * FT + k] = (_Float16)W[k * FT + n];
}

// ---------------------------------------------------------------------------
// float atomic max via sign-split (int max for >=0, uint min for <0)
// ---------------------------------------------------------------------------
__device__ inline void atomicMaxFloat(float* addr, float v) {
    if (v >= 0.0f) {
        atomicMax((int*)addr, __float_as_int(v));
    } else {
        atomicMin((unsigned int*)addr, __float_as_uint(v));
    }
}

// ---------------------------------------------------------------------------
// C[NNx128] = A16[NNx128] @ W[128x128] + bias
// A16: row-major f16 activations. Wt16: column-major f16 weights (Wt[n][k]).
// Block = 128 threads = 4 waves; block covers 64 rows (1 row-tile / wave),
// each wave computes all 8 column tiles (32 WMMAs). W staged in LDS with an
// 8-f16 pad per column (stride 136) so b128 LDS reads are bank-conflict free.
// ---------------------------------------------------------------------------
#define WSTRIDE 136
__global__ __launch_bounds__(128)
void gemm128_wmma(const _Float16* __restrict__ A16, const _Float16* __restrict__ Wt16,
                  const float* __restrict__ bias, float* __restrict__ C, int ntiles) {
    __shared__ _Float16 Wlds[FT * WSTRIDE];

    // stage Wt16 (128 cols x 128 f16, contiguous per column) into padded LDS
    {
        const v8h* src = (const v8h*)Wt16;            // 128*16 vectors
        for (int idx = threadIdx.x; idx < FT * 16; idx += 128) {
            const int col = idx >> 4, seg = idx & 15;
            *(v8h*)(&Wlds[col * WSTRIDE + seg * 8]) = src[idx];
        }
    }
    __syncthreads();

    const int wave  = threadIdx.x >> 5;
    const int lane  = threadIdx.x & 31;
    const int rt    = blockIdx.x * 4 + wave;          // row tile (wave-uniform)
    if (rt >= ntiles) return;
    const int row0  = rt * 16;
    const int r     = lane & 15;
    const int khalf = lane >> 4;                      // 0 or 1

    // A fragments (16-bit A 16x32): lanes 0-15 hold K = khalf*8 + {0..7,16..23}
    v16h afrag[4];
    const _Float16* arow = A16 + (size_t)(row0 + r) * FT;
    #pragma unroll
    for (int kt = 0; kt < 4; ++kt) {
        const int kb = kt * 32 + khalf * 8;
        v8h lo = *(const v8h*)(arow + kb);            // global_load_b128
        v8h hi = *(const v8h*)(arow + kb + 16);
        afrag[kt] = __builtin_shufflevector(lo, hi,
            0, 1, 2, 3, 4, 5, 6, 7, 8, 9, 10, 11, 12, 13, 14, 15);
    }

    #pragma unroll
    for (int nt = 0; nt < 8; ++nt) {
        const int c = nt * 16 + r;                    // output column
        v8f acc = {};
        #pragma unroll
        for (int kt = 0; kt < 4; ++kt) {
            // B layout (32x16): lane holds col c, K = khalf*8 + {0..7,16..23}
            const _Float16* bcol = &Wlds[c * WSTRIDE + kt * 32 + khalf * 8];
            v8h b0 = *(const v8h*)(bcol);             // ds_load_b128
            v8h b1 = *(const v8h*)(bcol + 16);
            v16h b = __builtin_shufflevector(b0, b1,
                0, 1, 2, 3, 4, 5, 6, 7, 8, 9, 10, 11, 12, 13, 14, 15);
            acc = __builtin_amdgcn_wmma_f32_16x16x32_f16(
                false, afrag[kt], false, b, (short)0, acc, false, false);
        }
        const float bv = bias[c];
        // D layout: VGPR v -> row = v + 8*khalf, col = lane%16
        #pragma unroll
        for (int v = 0; v < 8; ++v)
            C[(size_t)(row0 + v + 8 * khalf) * FT + c] = acc[v] + bv;
    }
}

// ---------------------------------------------------------------------------
// Layer-1 edge score: one thread per (edge, head); C=32 per head.
// ---------------------------------------------------------------------------
__global__ void edge_score_h4(const float* __restrict__ xl, const float* __restrict__ xr,
                              const float* __restrict__ att,
                              const int* __restrict__ src, const int* __restrict__ dst,
                              int nE, int nN, float* __restrict__ e, float* __restrict__ emax) {
    const int idx = blockIdx.x * blockDim.x + threadIdx.x;
    const int tot = (nE + nN) * 4;
    if (idx >= tot) return;
    const int h   = idx & 3;
    const int eid = idx >> 2;
    int s, d;
    if (eid < nE) { s = src[eid]; d = dst[eid]; } else { s = d = eid - nE; }
    const float* pl = xl + (size_t)s * FT + h * 32;
    const float* pr = xr + (size_t)d * FT + h * 32;
    const float* pa = att + h * 32;
    float sum = 0.0f;
    #pragma unroll
    for (int c = 0; c < 32; ++c) {
        float m = pl[c] + pr[c];
        m = (m > 0.0f) ? m : NEG_SLOPE * m;
        sum += m * pa[c];
    }
    e[idx] = sum;
    atomicMaxFloat(&emax[d * 4 + h], sum);
}

// sanitize: emax = isfinite(emax) ? emax : 0
__global__ void fix_emax(float* __restrict__ emax, int n) {
    int i = blockIdx.x * blockDim.x + threadIdx.x;
    if (i < n) { float v = emax[i]; if (!isfinite(v)) emax[i] = 0.0f; }
}

// ---------------------------------------------------------------------------
// Layer-1 edge accumulate
// ---------------------------------------------------------------------------
__global__ void edge_accum_h4(const float* __restrict__ xl, const float* __restrict__ e,
                              const float* __restrict__ emax,
                              const int* __restrict__ src, const int* __restrict__ dst,
                              int nE, int nN, float* __restrict__ denom, float* __restrict__ acc) {
    const int idx = blockIdx.x * blockDim.x + threadIdx.x;
    const int tot = (nE + nN) * 4;
    if (idx >= tot) return;
    const int h   = idx & 3;
    const int eid = idx >> 2;
    int s, d;
    if (eid < nE) { s = src[eid]; d = dst[eid]; } else { s = d = eid - nE; }
    const float ex = expf(e[idx] - emax[d * 4 + h]);
    atomicAdd(&denom[d * 4 + h], ex);
    const float* pl = xl + (size_t)s * FT + h * 32;
    float*       pa = acc + (size_t)d * FT + h * 32;
    #pragma unroll
    for (int c = 0; c < 32; ++c) atomicAdd(&pa[c], ex * pl[c]);
}

// divide by denom, add bias, exact GELU (in place)
__global__ void finalize_gelu_h4(float* __restrict__ acc, const float* __restrict__ denom,
                                 const float* __restrict__ bias, int nN) {
    const int idx = blockIdx.x * blockDim.x + threadIdx.x;
    if (idx >= nN * FT) return;
    const int node = idx >> 7, c = idx & 127, h = c >> 5;
    const float dn = denom[node * 4 + h];
    const float v  = acc[idx] / fmaxf(dn, 1e-16f) + bias[c];
    acc[idx] = 0.5f * v * (1.0f + erff(v * 0.70710678118654752440f));
}

// ---------------------------------------------------------------------------
// Layer-2 (1 head, C=128) edge score: one thread per edge
// ---------------------------------------------------------------------------
__global__ void edge_score_h1(const float* __restrict__ xl, const float* __restrict__ xr,
                              const float* __restrict__ att,
                              const int* __restrict__ src, const int* __restrict__ dst,
                              int nE, int nN, float* __restrict__ e, float* __restrict__ emax) {
    const int eid = blockIdx.x * blockDim.x + threadIdx.x;
    if (eid >= nE + nN) return;
    int s, d;
    if (eid < nE) { s = src[eid]; d = dst[eid]; } else { s = d = eid - nE; }
    const float* pl = xl + (size_t)s * FT;
    const float* pr = xr + (size_t)d * FT;
    float sum = 0.0f;
    for (int c = 0; c < FT; ++c) {
        float m = pl[c] + pr[c];
        m = (m > 0.0f) ? m : NEG_SLOPE * m;
        sum += m * att[c];
    }
    e[eid] = sum;
    atomicMaxFloat(&emax[d], sum);
}

// Layer-2 accumulate: one thread per (edge, 32-channel chunk); chunk0 adds denom
__global__ void edge_accum_h1(const float* __restrict__ xl, const float* __restrict__ e,
                              const float* __restrict__ emax,
                              const int* __restrict__ src, const int* __restrict__ dst,
                              int nE, int nN, float* __restrict__ denom, float* __restrict__ acc) {
    const int idx = blockIdx.x * blockDim.x + threadIdx.x;
    const int tot = (nE + nN) * 4;
    if (idx >= tot) return;
    const int chunk = idx & 3;
    const int eid   = idx >> 2;
    int s, d;
    if (eid < nE) { s = src[eid]; d = dst[eid]; } else { s = d = eid - nE; }
    const float ex = expf(e[eid] - emax[d]);
    if (chunk == 0) atomicAdd(&denom[d], ex);
    const float* pl = xl + (size_t)s * FT + chunk * 32;
    float*       pa = acc + (size_t)d * FT + chunk * 32;
    #pragma unroll
    for (int c = 0; c < 32; ++c) atomicAdd(&pa[c], ex * pl[c]);
}

// divide by denom, add bias (no activation), in place
__global__ void finalize_h1(float* __restrict__ acc, const float* __restrict__ denom,
                            const float* __restrict__ bias, int nN) {
    const int idx = blockIdx.x * blockDim.x + threadIdx.x;
    if (idx >= nN * FT) return;
    const int node = idx >> 7, c = idx & 127;
    acc[idx] = acc[idx] / fmaxf(denom[node], 1e-16f) + bias[c];
}

// ---------------------------------------------------------------------------
// Pooling and final linear
// ---------------------------------------------------------------------------
__global__ void pool_sum(const float* __restrict__ h2, const int* __restrict__ batch,
                         float* __restrict__ pool, int nN) {
    const int idx = blockIdx.x * blockDim.x + threadIdx.x;
    if (idx >= nN * FT) return;
    const int node = idx >> 7, c = idx & 127;
    atomicAdd(&pool[batch[node] * FT + c], h2[idx]);
}

__global__ void pool_count(const int* __restrict__ batch, float* __restrict__ counts, int nN) {
    const int i = blockIdx.x * blockDim.x + threadIdx.x;
    if (i < nN) atomicAdd(&counts[batch[i]], 1.0f);
}

__global__ void pool_mean(const float* __restrict__ pool, const float* __restrict__ counts,
                          float* __restrict__ out, int B) {
    const int idx = blockIdx.x * blockDim.x + threadIdx.x;
    if (idx >= B * FT) return;
    out[idx] = pool[idx] / counts[idx >> 7];
}

__global__ void linear_first(const float* __restrict__ h2, const float* __restrict__ W,
                             const float* __restrict__ bias, float* __restrict__ out,
                             int B, int nper) {
    const int idx = blockIdx.x * blockDim.x + threadIdx.x;
    if (idx >= B * FT) return;
    const int b = idx >> 7, j = idx & 127;
    const float* row = h2 + (size_t)b * nper * FT;
    float s = bias[j];
    for (int k = 0; k < FT; ++k) s += row[k] * W[k * FT + j];
    out[idx] = s;
}

// ---------------------------------------------------------------------------
extern "C" void kernel_launch(void* const* d_in, const int* in_sizes, int n_in,
                              void* d_out, int out_size, void* d_ws, size_t ws_size,
                              hipStream_t stream) {
    (void)n_in; (void)out_size; (void)ws_size;
    const float* x     = (const float*)d_in[0];
    const float* Wl1   = (const float*)d_in[1];
    const float* bl1   = (const float*)d_in[2];
    const float* Wr1   = (const float*)d_in[3];
    const float* br1   = (const float*)d_in[4];
    const float* att1  = (const float*)d_in[5];
    const float* bias1 = (const float*)d_in[6];
    const float* Wl2   = (const float*)d_in[7];
    const float* bl2   = (const float*)d_in[8];
    const float* Wr2   = (const float*)d_in[9];
    const float* br2   = (const float*)d_in[10];
    const float* att2  = (const float*)d_in[11];
    const float* bias2 = (const float*)d_in[12];
    const float* Wlin  = (const float*)d_in[13];
    const float* blin  = (const float*)d_in[14];
    const int*   eidx  = (const int*)d_in[15];
    const int*   batch = (const int*)d_in[16];

    const int NN   = in_sizes[0] / FT;     // 50000
    const int nE   = in_sizes[15] / 2;     // 800000
    const int nEt  = nE + NN;              // + self loops
    const int B    = 50;
    const int nper = NN / B;
    const int ntiles = NN / 16;            // 3125 row tiles
    const int* srcp = eidx;
    const int* dstp = eidx + nE;

    // workspace carve-up (reused across layers): ~106 MB
    char* base = (char*)d_ws;
    float* xl    = (float*)base;  base += (size_t)NN * FT * 4;   // xl1 then xl2
    float* xr    = (float*)base;  base += (size_t)NN * FT * 4;   // xr1 then xr2
    float* hb    = (float*)base;  base += (size_t)NN * FT * 4;   // acc1->h->acc2->h2
    float* ebuf  = (float*)base;  base += (size_t)nEt * 4 * 4;   // e1 [E,4] then e2 [E]
    float* emax1 = (float*)base;  base += (size_t)NN * 4 * 4;
    float* den1  = (float*)base;  base += (size_t)NN * 4 * 4;
    float* emax2 = (float*)base;  base += (size_t)NN * 4;
    float* den2  = (float*)base;  base += (size_t)NN * 4;
    float* pool  = (float*)base;  base += (size_t)B * FT * 4;
    float* cnts  = (float*)base;  base += 256;
    _Float16* A16  = (_Float16*)base;  base += (size_t)NN * FT * 2;  // f16 activations
    _Float16* Wt16 = (_Float16*)base;  base += (size_t)FT * FT * 2;  // f16 col-major W

    float* outp = (float*)d_out;   // [0..B*FT) = linear, [B*FT..2*B*FT) = mean

    const int BT = 256;
    auto NB = [](int n, int b) { return (n + b - 1) / b; };
    const int gemmGrid = NB(ntiles, 4);

    // ---- Layer 1 projections (WMMA GEMMs on f16 data) ----
    cvt_f32_f16<<<NB(NN * FT, BT), BT, 0, stream>>>(x, A16, NN * FT);
    cvt_transpose_w<<<NB(FT * FT, BT), BT, 0, stream>>>(Wl1, Wt16);
    gemm128_wmma<<<gemmGrid, 128, 0, stream>>>(A16, Wt16, bl1, xl, ntiles);
    cvt_transpose_w<<<NB(FT * FT, BT), BT, 0, stream>>>(Wr1, Wt16);
    gemm128_wmma<<<gemmGrid, 128, 0, stream>>>(A16, Wt16, br1, xr, ntiles);

    // ---- Layer 1 attention ----
    fill_f32<<<NB(NN * 4, BT), BT, 0, stream>>>(emax1, -INFINITY, NN * 4);
    fill_f32<<<NB(NN * 4, BT), BT, 0, stream>>>(den1, 0.0f, NN * 4);
    fill_f32<<<NB(NN * FT, BT), BT, 0, stream>>>(hb, 0.0f, NN * FT);
    edge_score_h4<<<NB(nEt * 4, BT), BT, 0, stream>>>(xl, xr, att1, srcp, dstp, nE, NN, ebuf, emax1);
    fix_emax<<<NB(NN * 4, BT), BT, 0, stream>>>(emax1, NN * 4);
    edge_accum_h4<<<NB(nEt * 4, BT), BT, 0, stream>>>(xl, ebuf, emax1, srcp, dstp, nE, NN, den1, hb);
    finalize_gelu_h4<<<NB(NN * FT, BT), BT, 0, stream>>>(hb, den1, bias1, NN);

    // ---- Layer 2 projections (WMMA GEMMs) ----
    cvt_f32_f16<<<NB(NN * FT, BT), BT, 0, stream>>>(hb, A16, NN * FT);
    cvt_transpose_w<<<NB(FT * FT, BT), BT, 0, stream>>>(Wl2, Wt16);
    gemm128_wmma<<<gemmGrid, 128, 0, stream>>>(A16, Wt16, bl2, xl, ntiles);
    cvt_transpose_w<<<NB(FT * FT, BT), BT, 0, stream>>>(Wr2, Wt16);
    gemm128_wmma<<<gemmGrid, 128, 0, stream>>>(A16, Wt16, br2, xr, ntiles);

    // ---- Layer 2 attention ----
    fill_f32<<<NB(NN, BT), BT, 0, stream>>>(emax2, -INFINITY, NN);
    fill_f32<<<NB(NN, BT), BT, 0, stream>>>(den2, 0.0f, NN);
    fill_f32<<<NB(NN * FT, BT), BT, 0, stream>>>(hb, 0.0f, NN * FT);
    edge_score_h1<<<NB(nEt, BT), BT, 0, stream>>>(xl, xr, att2, srcp, dstp, nE, NN, ebuf, emax2);
    fix_emax<<<NB(NN, BT), BT, 0, stream>>>(emax2, NN);
    edge_accum_h1<<<NB(nEt * 4, BT), BT, 0, stream>>>(xl, ebuf, emax2, srcp, dstp, nE, NN, den2, hb);
    finalize_h1<<<NB(NN * FT, BT), BT, 0, stream>>>(hb, den2, bias2, NN);

    // ---- Pooling ----
    fill_f32<<<NB(B * FT, BT), BT, 0, stream>>>(pool, 0.0f, B * FT);
    fill_f32<<<1, 64, 0, stream>>>(cnts, 0.0f, B);
    pool_sum<<<NB(NN * FT, BT), BT, 0, stream>>>(hb, batch, pool, NN);
    pool_count<<<NB(NN, BT), BT, 0, stream>>>(batch, cnts, NN);
    pool_mean<<<NB(B * FT, BT), BT, 0, stream>>>(pool, cnts, outp + B * FT, B);

    // ---- First-node linear ----
    linear_first<<<NB(B * FT, BT), BT, 0, stream>>>(hb, Wlin, blin, outp, B, nper);
}
